// BasicBlock_38903813767750
// MI455X (gfx1250) — 8-bit matrix operand order repaired and checked by reading, compiled, not run on hardware
//
#include <hip/hip_runtime.h>
#include <cmath>

typedef int v8i __attribute__((ext_vector_type(8)));

#define B_    32
#define C_    256
#define H_    56
#define W_    56
#define HW    (H_*W_)
#define NPIX  (B_*HW)
#define NELEM ((size_t)B_*C_*HW)
#define KTOT  (C_*9)
#define BN_EPS 1e-5f

#define XS_BYTES ((size_t)NPIX*C_)
#define WP_OFF   XS_BYTES
#define WP_BYTES ((size_t)16*9*4*1024)
#define SC_OFF   (WP_OFF + WP_BYTES)
#define ST_OFF   (SC_OFF + 1024)

__global__ __launch_bounds__(256) void pack_x(const float* __restrict__ x,
                                              signed char* __restrict__ xs) {
    int gid = blockIdx.x * 256 + threadIdx.x;
    int p   = gid >> 6;
    int c4  = (gid & 63) << 2;
    int b   = p / HW, hw = p - b * HW;
    const float* xp = x + (size_t)b * C_ * HW + hw;
    unsigned out = 0;
#pragma unroll
    for (int j = 0; j < 4; ++j) {
        float v = xp[(size_t)(c4 + j) * HW];
        int s = (v > 0.f) - (v < 0.f);
        out |= (unsigned)(s & 0xff) << (8 * j);
    }
    *(unsigned*)(xs + (size_t)p * C_ + c4) = out;
}

__global__ __launch_bounds__(256) void wscale(const float* __restrict__ w,
                                              float* __restrict__ scale) {
    __shared__ float red[256];
    int co = blockIdx.x;
    float s = 0.f;
    for (int i = threadIdx.x; i < KTOT; i += 256)
        s += fabsf(w[(size_t)co * KTOT + i]);
    red[threadIdx.x] = s;
    __syncthreads();
    for (int off = 128; off > 0; off >>= 1) {
        if (threadIdx.x < off) red[threadIdx.x] += red[threadIdx.x + off];
        __syncthreads();
    }
    if (threadIdx.x == 0) scale[co] = red[0] / (float)KTOT;
}

__global__ __launch_bounds__(256) void pack_w(const float* __restrict__ w,
                                              signed char* __restrict__ wp) {
    int t = blockIdx.x * 256 + threadIdx.x;
    int frag = t >> 10;
    int r    = t & 1023;
    int lane = r >> 5;
    int o    = r & 31;
    int v = o >> 2, j = o & 3;
    int kc  = frag & 3;
    int tap = (frag >> 2) % 9;
    int ct  = frag / 36;
    int co  = ct * 16 + (lane & 15);
    int ci  = kc * 64 + (lane >> 4) * 8 + (v & 1) * 4 + (v >> 1) * 16 + j;
    int kh = tap / 3, kw = tap % 3;
    float wv = w[(((size_t)co * C_ + ci) * 3 + kh) * 3 + kw];
    wp[t] = (signed char)((wv > 0.f) - (wv < 0.f));
}

#define LROW   (66*64)
#define STRIPE (3*LROW)
__global__ __launch_bounds__(256) void conv_wmma(const signed char* __restrict__ xs,
                                                 const signed char* __restrict__ wp,
                                                 const float* __restrict__ scale,
                                                 float* __restrict__ out) {
    __shared__ signed char lds[2 * STRIPE];
    int bid = blockIdx.x;
    int b = bid / H_, h = bid % H_;
    int tid  = threadIdx.x;
    int lane = tid & 31, wave = tid >> 5;
    int m_tile = wave & 3, nhalf = wave >> 2;
    int half = lane >> 4;

    v8i acc[8] = {};
    const int4* wp4 = (const int4*)wp;

    auto stage = [&](int kc, int buf) {
        int dstbase = buf * STRIPE;
        for (int i = tid; i < 792; i += 256) {
            int off  = i << 4;
            int row  = off / LROW;
            int rem  = off - row * LROW;
            int px   = rem >> 6;
            int coff = rem & 63;
            int hh = h - 1 + row;
            int ww = px - 1;
            if (hh >= 0 && hh < H_ && ww >= 0 && ww < W_) {
                unsigned ldsa = (unsigned)(size_t)(lds + dstbase + off);
                unsigned goff = (unsigned)(((b * H_ + hh) * W_ + ww) * C_
                                           + kc * 64 + coff);
                asm volatile("global_load_async_to_lds_b128 %0, %1, %2"
                             :: "v"(ldsa), "v"(goff), "s"(xs)
                             : "memory");
            } else {
                *(int4*)(lds + dstbase + off) = (int4){0, 0, 0, 0};
            }
        }
    };

    stage(0, 0);
    asm volatile("s_wait_asynccnt 0x0" ::: "memory");
    __syncthreads();

    for (int kc = 0; kc < 4; ++kc) {
        int buf = kc & 1;
        if (kc < 3) stage(kc + 1, buf ^ 1);
        const signed char* cur = lds + buf * STRIPE;

        for (int tap = 0; tap < 9; ++tap) {
            int kh = tap / 3, kw = tap % 3;
            int abase = (kh * 66 + m_tile * 16 + (lane & 15) + kw) * 64 + half * 8;
            int2 p0 = *(const int2*)(cur + abase);
            int2 p1 = *(const int2*)(cur + abase + 16);
            int2 p2 = *(const int2*)(cur + abase + 32);
            int2 p3 = *(const int2*)(cur + abase + 48);
            v8i a;
            a[0] = p0.x; a[1] = p0.y; a[2] = p1.x; a[3] = p1.y;
            a[4] = p2.x; a[5] = p2.y; a[6] = p3.x; a[7] = p3.y;

#pragma unroll
            for (int t = 0; t < 8; ++t) {
                int ct   = nhalf * 8 + t;
                int fidx = ((ct * 9 + tap) * 4 + kc) * 64 + lane * 2;
                if (t == 0 && tap < 8)
                    __builtin_prefetch(&wp4[((ct * 9 + tap + 1) * 4 + kc) * 64 + lane * 2], 0, 3);
                int4 b0 = wp4[fidx];
                int4 b1 = wp4[fidx + 1];
                v8i bb;
                bb[0] = b0.x; bb[1] = b0.y; bb[2] = b0.z; bb[3] = b0.w;
                bb[4] = b1.x; bb[5] = b1.y; bb[6] = b1.z; bb[7] = b1.w;
                acc[t] = __builtin_amdgcn_wmma_i32_16x16x64_iu8(
                    true, a, true, bb, acc[t], false, false);
            }
        }
        asm volatile("s_wait_asynccnt 0x0" ::: "memory");
        __syncthreads();
    }

    int base_w = m_tile * 16 + half * 8;
    if (base_w < W_) {
#pragma unroll
        for (int t = 0; t < 8; ++t) {
            int co = (nhalf * 8 + t) * 16 + (lane & 15);
            float s = scale[co];
            float4 f0, f1;
            f0.x = s * (float)acc[t][0]; f0.y = s * (float)acc[t][1];
            f0.z = s * (float)acc[t][2]; f0.w = s * (float)acc[t][3];
            f1.x = s * (float)acc[t][4]; f1.y = s * (float)acc[t][5];
            f1.z = s * (float)acc[t][6]; f1.w = s * (float)acc[t][7];
            float* ptr = out + ((size_t)(b * C_ + co) * H_ + h) * W_ + base_w;
            *(float4*)ptr       = f0;
            *(float4*)(ptr + 4) = f1;
        }
    }
}

__global__ __launch_bounds__(256) void bn_stats(const float* __restrict__ y,
                                                float* __restrict__ stats) {
    __shared__ double red[512];
    int c = blockIdx.x;
    double s = 0.0, sq = 0.0;
    for (int i = threadIdx.x; i < NPIX; i += 256) {
        int b = i / HW, p = i - b * HW;
        float v = y[((size_t)b * C_ + c) * HW + p];
        s  += (double)v;
        sq += (double)v * (double)v;
    }
    red[threadIdx.x]       = s;
    red[256 + threadIdx.x] = sq;
    __syncthreads();
    for (int off = 128; off > 0; off >>= 1) {
        if (threadIdx.x < off) {
            red[threadIdx.x]       += red[threadIdx.x + off];
            red[256 + threadIdx.x] += red[256 + threadIdx.x + off];
        }
        __syncthreads();
    }
    if (threadIdx.x == 0) {
        double mean = red[0] / (double)NPIX;
        double var  = red[256] / (double)NPIX - mean * mean;
        stats[c]       = (float)mean;
        stats[256 + c] = (float)var;
    }
}

__global__ __launch_bounds__(256) void bn_relu(float* __restrict__ y,
                                               const float* __restrict__ stats,
                                               const float* __restrict__ gamma,
                                               const float* __restrict__ beta) {
    size_t gid = (size_t)blockIdx.x * 256 + threadIdx.x;
    size_t idx = gid * 4;
    int c = (int)((idx / HW) % C_);
    float mean = stats[c], var = stats[256 + c];
    float inv = rsqrtf(var + BN_EPS);
    float g = gamma[c] * inv;
    float bb = beta[c] - mean * g;
    float4 v = *(float4*)(y + idx);
    v.x = fmaxf(v.x * g + bb, 0.f);
    v.y = fmaxf(v.y * g + bb, 0.f);
    v.z = fmaxf(v.z * g + bb, 0.f);
    v.w = fmaxf(v.w * g + bb, 0.f);
    *(float4*)(y + idx) = v;
}

extern "C" void kernel_launch(void* const* d_in, const int* in_sizes, int n_in,
                              void* d_out, int out_size, void* d_ws, size_t ws_size,
                              hipStream_t stream) {
    (void)in_sizes; (void)n_in; (void)out_size; (void)ws_size;
    const float* x     = (const float*)d_in[0];
    const float* w     = (const float*)d_in[1];
    const float* gamma = (const float*)d_in[2];
    const float* beta  = (const float*)d_in[3];
    float* out = (float*)d_out;

    signed char* xs  = (signed char*)d_ws;
    signed char* wpk = (signed char*)d_ws + WP_OFF;
    float* scale     = (float*)((char*)d_ws + SC_OFF);
    float* stats     = (float*)((char*)d_ws + ST_OFF);

    pack_x   <<<(NPIX * 64) / 256, 256, 0, stream>>>(x, xs);
    wscale   <<<C_,                256, 0, stream>>>(w, scale);
    pack_w   <<<(int)(WP_BYTES/256),256, 0, stream>>>(w, wpk);
    conv_wmma<<<B_ * H_,           256, 0, stream>>>(xs, wpk, scale, out);
    bn_stats <<<C_,                256, 0, stream>>>(out, stats);
    bn_relu  <<<(int)(NELEM/4/256),256, 0, stream>>>(out, stats, gamma, beta);
}
